// TLSTM_56624848830838
// MI455X (gfx1250) — matrix operand order repaired and checked by reading, not compiled, not run on hardware
//
#include <hip/hip_runtime.h>
#include <hip/hip_bf16.h>

// ---------------------------------------------------------------------------
// T-LSTM on MI455X (gfx1250): all GEMMs via v_wmma_f32_16x16x32_bf16.
// B=256 T=256 D=256 H=512 FC=256 O=2.
// Recurrent scan (172 GFLOP over 256 serial steps) dominates; bf16 weights
// (~5MB) live in L2 across all steps, so WMMA issue rate / step latency is
// the limiter, not HBM (23.3 TB/s).  Each wave owns TWO 16x16 batch tiles of
// one hidden tile so every weight (B) fragment feeds two WMMAs:
// h/c loop = 18 b128 loads : 10 WMMAs (vs 14:5 in the 1-tile version).
// ---------------------------------------------------------------------------

typedef __bf16 bf16_t;
typedef __attribute__((ext_vector_type(16))) __bf16 v16bf;
typedef __attribute__((ext_vector_type(8)))  __bf16 v8bf;
typedef __attribute__((ext_vector_type(8)))  float  v8f;

namespace {
constexpr int kB  = 256;
constexpr int kT  = 256;
constexpr int kD  = 256;
constexpr int kH  = 512;
constexpr int kFC = 256;
constexpr int kO  = 2;
}

__device__ __forceinline__ bf16_t f2bf(float f) {
  union { float f; unsigned u; } v; v.f = f;
  unsigned r = v.u + 0x7FFFu + ((v.u >> 16) & 1u);          // RNE
  union { unsigned short s; bf16_t b; } o; o.s = (unsigned short)(r >> 16);
  return o.b;
}

__device__ __forceinline__ float fsigm(float x) {
  return 1.0f / (1.0f + __expf(-x));
}
__device__ __forceinline__ float ftanh(float x) {
  x = fminf(fmaxf(x, -20.0f), 20.0f);
  float e = __expf(-2.0f * x);
  return (1.0f - e) / (1.0f + e);
}

// A fragment (16x32 bf16, M x K): lanes 0-15 -> K 0..15,
// lanes 16-31 -> K 16..31.  p already includes +16 for hi half.
__device__ __forceinline__ v16bf ldA(const bf16_t* p) {
  v8bf lo  = *(const v8bf*)(p);
  v8bf hi8 = *(const v8bf*)(p + 8);
  return __builtin_shufflevector(lo, hi8, 0,1,2,3,4,5,6,7,8,9,10,11,12,13,14,15);
}
// B fragment (32x16 bf16, K x N), B[k][n] = W[ntile+n][k]: lane n holds row
// (ntile+n) of W, 16 contiguous k values (lanes 16-31 use k+16) -> 32B load.
__device__ __forceinline__ v16bf ldB(const bf16_t* p) { return *(const v16bf*)p; }

__device__ __forceinline__ v8f wmma_bf16(v16bf A, v16bf Bm, v8f C) {
  return __builtin_amdgcn_wmma_f32_16x16x32_bf16(
      /*neg_a=*/false, A, /*neg_b=*/false, Bm,
      /*c_mod=*/(short)0, C, /*reuse_a=*/false, /*reuse_b=*/false);
}

// ------------------------------- conversion --------------------------------

__global__ void tlstm_cvt_bf16(const float* __restrict__ s,
                               bf16_t* __restrict__ d, int n) {
  int i = blockIdx.x * blockDim.x + threadIdx.x;
  int st = gridDim.x * blockDim.x;
  for (; i < n; i += st) d[i] = f2bf(s[i]);
}

__global__ void tlstm_zero_state(bf16_t* __restrict__ hb, bf16_t* __restrict__ cb,
                                 float* __restrict__ c32, int n) {
  int i = blockIdx.x * blockDim.x + threadIdx.x;
  int st = gridDim.x * blockDim.x;
  bf16_t z = f2bf(0.0f);
  for (; i < n; i += st) { hb[i] = z; cb[i] = z; c32[i] = 0.0f; }
}

// ------------------------------- step kernel -------------------------------

struct StepArgs {
  const bf16_t* xb;     // [B,T,D] bf16
  const float*  timep;  // [B,T]
  const bf16_t* Wx;     // 4 x [H,D] bf16 : Wi,Wf,Wo,Wc
  const bf16_t* Uw;     // 5 x [H,H] bf16 : Ui,Uf,Uo,Uc,Wd
  const float*  Wi_b; const float* bi;
  const float*  Wf_b; const float* bfg;
  const float*  Wo_b; const float* bo;
  const float*  Wc_b; const float* bc;
  const float*  Ui_b; const float* Uf_b; const float* Uo_b; const float* Uc_b;
  const float*  b_decomp;
  bf16_t* hbuf0; bf16_t* hbuf1;
  bf16_t* cbuf0; bf16_t* cbuf1;
  float*  c32;          // [B,H] f32 running cell state
};

// 256 waves: 8 batch-tile-pairs x 32 hidden-tiles.  Each wave computes the
// 5 gate GEMMs for a 32x16 (b,h) block (two 16x16 WMMA tiles sharing every
// weight fragment), then does the elementwise T-LSTM update in-register.
__global__ __launch_bounds__(256) void tlstm_step(StepArgs a, int t, int pr) {
  const int lane = threadIdx.x & 31;
  const int wid  = (blockIdx.x * blockDim.x + threadIdx.x) >> 5;
  const int nh   = wid & 31;          // hidden tile   (0..31)
  const int mg   = wid >> 5;          // batch tile pair (0..7)
  const int l15  = lane & 15;
  const int lhi  = lane >> 4;
  const int g8   = lhi * 16;          // A-fragment k sub-offset
  const int kb16 = lhi * 16;          // B-fragment k sub-offset
  const int arow0 = mg * 32 + l15;    // batch rows this lane loads (A)
  const int arow1 = arow0 + 16;
  const int nrow  = nh * 16 + l15;    // weight row this lane loads (B)

  const bf16_t* hprev = pr ? a.hbuf1 : a.hbuf0;
  const bf16_t* cprev = pr ? a.cbuf1 : a.cbuf0;
  bf16_t* hnext = pr ? a.hbuf0 : a.hbuf1;
  bf16_t* cnext = pr ? a.cbuf0 : a.cbuf1;

  v8f ai0{}, af0{}, ao0{}, ag0{}, ad0{};   // batch tile 0 accumulators
  v8f ai1{}, af1{}, ao1{}, ag1{}, ad1{};   // batch tile 1 accumulators

  // ---- x @ Wg^T contribution (K = D = 256) ----
  const bf16_t* xr0 = a.xb + ((size_t)arow0 * kT + t) * kD + g8;
  const bf16_t* xr1 = a.xb + ((size_t)arow1 * kT + t) * kD + g8;
  const bf16_t* wi = a.Wx + (size_t)0 * kH * kD + (size_t)nrow * kD + kb16;
  const bf16_t* wf = a.Wx + (size_t)1 * kH * kD + (size_t)nrow * kD + kb16;
  const bf16_t* wo = a.Wx + (size_t)2 * kH * kD + (size_t)nrow * kD + kb16;
  const bf16_t* wc = a.Wx + (size_t)3 * kH * kD + (size_t)nrow * kD + kb16;
#pragma unroll
  for (int k = 0; k < kD; k += 32) {
    v16bf a0 = ldA(xr0 + k);
    v16bf a1 = ldA(xr1 + k);
    v16bf b;
    b = ldB(wi + k); ai0 = wmma_bf16(a0, b, ai0); ai1 = wmma_bf16(a1, b, ai1);
    b = ldB(wf + k); af0 = wmma_bf16(a0, b, af0); af1 = wmma_bf16(a1, b, af1);
    b = ldB(wo + k); ao0 = wmma_bf16(a0, b, ao0); ao1 = wmma_bf16(a1, b, ao1);
    b = ldB(wc + k); ag0 = wmma_bf16(a0, b, ag0); ag1 = wmma_bf16(a1, b, ag1);
  }

  // ---- h @ Ug^T and c @ Wd^T contribution (K = H = 512) ----
  const bf16_t* hr0 = hprev + (size_t)arow0 * kH + g8;
  const bf16_t* hr1 = hprev + (size_t)arow1 * kH + g8;
  const bf16_t* cr0 = cprev + (size_t)arow0 * kH + g8;
  const bf16_t* cr1 = cprev + (size_t)arow1 * kH + g8;
  const bf16_t* ui = a.Uw + (size_t)0 * kH * kH + (size_t)nrow * kH + kb16;
  const bf16_t* uf = a.Uw + (size_t)1 * kH * kH + (size_t)nrow * kH + kb16;
  const bf16_t* uo = a.Uw + (size_t)2 * kH * kH + (size_t)nrow * kH + kb16;
  const bf16_t* uc = a.Uw + (size_t)3 * kH * kH + (size_t)nrow * kH + kb16;
  const bf16_t* wd = a.Uw + (size_t)4 * kH * kH + (size_t)nrow * kH + kb16;
#pragma unroll
  for (int k = 0; k < kH; k += 32) {
    v16bf h0 = ldA(hr0 + k);
    v16bf h1 = ldA(hr1 + k);
    v16bf c0 = ldA(cr0 + k);
    v16bf c1 = ldA(cr1 + k);
    v16bf b;
    b = ldB(ui + k); ai0 = wmma_bf16(h0, b, ai0); ai1 = wmma_bf16(h1, b, ai1);
    b = ldB(uf + k); af0 = wmma_bf16(h0, b, af0); af1 = wmma_bf16(h1, b, af1);
    b = ldB(uo + k); ao0 = wmma_bf16(h0, b, ao0); ao1 = wmma_bf16(h1, b, ao1);
    b = ldB(uc + k); ag0 = wmma_bf16(h0, b, ag0); ag1 = wmma_bf16(h1, b, ag1);
    b = ldB(wd + k); ad0 = wmma_bf16(c0, b, ad0); ad1 = wmma_bf16(c1, b, ad1);
  }

  // ---- elementwise T-LSTM update (C/D layout: lane holds rows r(+8), col=lane&15)
  const int hcol = nh * 16 + l15;
  const float uib = a.Ui_b[hcol], ufb = a.Uf_b[hcol];
  const float uob = a.Uo_b[hcol], ucb = a.Uc_b[hcol];
  const float bdc = a.b_decomp[hcol];
  const float xbi = a.Wi_b[hcol] + a.bi[hcol];
  const float xbf = a.Wf_b[hcol] + a.bfg[hcol];
  const float xbo = a.Wo_b[hcol] + a.bo[hcol];
  const float xbc = a.Wc_b[hcol] + a.bc[hcol];

#pragma unroll
  for (int half = 0; half < 2; ++half) {
    const int mbase = mg * 32 + half * 16 + lhi * 8;
    const v8f& vi = half ? ai1 : ai0;
    const v8f& vf = half ? af1 : af0;
    const v8f& vo = half ? ao1 : ao0;
    const v8f& vg = half ? ag1 : ag0;
    const v8f& vd = half ? ad1 : ad0;
#pragma unroll
    for (int r = 0; r < 8; ++r) {
      const int b   = mbase + r;
      const int idx = b * kH + hcol;
      const float tv   = a.timep[b * kT + t];
      const float Tt   = 1.0f / __logf(tv + 2.7183f);
      const float cold = a.c32[idx];
      const float cst  = ftanh(vd[r] + bdc);
      const float c1   = cold - cst + Tt * cst;
      const float iv   = fsigm(vi[r] + xbi + uib);
      const float fv   = fsigm(vf[r] + xbf + ufb);
      const float ov   = fsigm(vo[r] + xbo + uob);
      const float ch   = ftanh(vg[r] + xbc + ucb);
      const float cn   = fv * c1 + iv * ch;
      const float hn   = ov * ftanh(cn);
      a.c32[idx]  = cn;
      cnext[idx]  = f2bf(cn);
      hnext[idx]  = f2bf(hn);
    }
  }
}

// ------------------------------ classifier ---------------------------------

// fc_out = relu(h_T @ fc_w^T + fc_b) : M=256, N=256, K=512 (WMMA bf16)
__global__ __launch_bounds__(256) void tlstm_fc(const bf16_t* __restrict__ hb,
                                                const bf16_t* __restrict__ fcw,
                                                const float* __restrict__ fcb,
                                                float* __restrict__ out) {
  const int lane = threadIdx.x & 31;
  const int wid  = (blockIdx.x * blockDim.x + threadIdx.x) >> 5;
  const int nt   = wid & 15;
  const int mt   = wid >> 4;
  const int l15  = lane & 15;
  const int lhi  = lane >> 4;
  const int arow = mt * 16 + l15;
  const int nrow = nt * 16 + l15;

  const bf16_t* ha = hb  + (size_t)arow * kH + lhi * 16;
  const bf16_t* wb = fcw + (size_t)nrow * kH + lhi * 16;
  v8f acc{};
#pragma unroll
  for (int k = 0; k < kH; k += 32)
    acc = wmma_bf16(ldA(ha + k), ldB(wb + k), acc);

  const int col   = nt * 16 + l15;
  const int mbase = mt * 16 + lhi * 8;
  const float bias = fcb[col];
#pragma unroll
  for (int r = 0; r < 8; ++r) {
    float v = acc[r] + bias;
    out[(mbase + r) * kFC + col] = v > 0.0f ? v : 0.0f;
  }
}

// out = fc_out @ cls_w^T + cls_b : M=256, K=256, N=2 (tiny; plain dots)
__global__ void tlstm_cls(const float* __restrict__ fc_out,
                          const float* __restrict__ clsw,
                          const float* __restrict__ clsb,
                          float* __restrict__ out) {
  int g = blockIdx.x * blockDim.x + threadIdx.x;   // [0, B*O)
  if (g >= kB * kO) return;
  int b = g >> 1, o = g & 1;
  float s = clsb[o];
  const float* fr = fc_out + (size_t)b * kFC;
  const float* wr = clsw + (size_t)o * kFC;
#pragma unroll 8
  for (int f = 0; f < kFC; ++f) s += fr[f] * wr[f];
  out[g] = s;
}

// ------------------------------ launcher -----------------------------------

extern "C" void kernel_launch(void* const* d_in, const int* in_sizes, int n_in,
                              void* d_out, int out_size, void* d_ws, size_t ws_size,
                              hipStream_t stream) {
  (void)in_sizes; (void)n_in; (void)out_size; (void)ws_size;

  // inputs (setup_inputs order)
  const float* x      = (const float*)d_in[0];
  const float* timep  = (const float*)d_in[1];
  const float* Wi_w   = (const float*)d_in[2];
  const float* Wi_b   = (const float*)d_in[3];
  const float* Ui_w   = (const float*)d_in[4];
  const float* Ui_b   = (const float*)d_in[5];
  const float* bi     = (const float*)d_in[6];
  const float* Wf_w   = (const float*)d_in[7];
  const float* Wf_b   = (const float*)d_in[8];
  const float* Uf_w   = (const float*)d_in[9];
  const float* Uf_b   = (const float*)d_in[10];
  const float* bfg    = (const float*)d_in[11];
  const float* Wo_w   = (const float*)d_in[12];
  const float* Wo_b   = (const float*)d_in[13];
  const float* Uo_w   = (const float*)d_in[14];
  const float* Uo_b   = (const float*)d_in[15];
  const float* bo     = (const float*)d_in[16];
  const float* Wc_w   = (const float*)d_in[17];
  const float* Wc_b   = (const float*)d_in[18];
  const float* Uc_w   = (const float*)d_in[19];
  const float* Uc_b   = (const float*)d_in[20];
  const float* bc     = (const float*)d_in[21];
  const float* Wd_w   = (const float*)d_in[22];
  const float* b_dec  = (const float*)d_in[23];
  const float* fc_w   = (const float*)d_in[24];
  const float* fc_b   = (const float*)d_in[25];
  const float* cls_w  = (const float*)d_in[26];
  const float* cls_b  = (const float*)d_in[27];

  // workspace layout (256B aligned slabs, ~37.5 MB total)
  char* ws = (char*)d_ws;
  size_t off = 0;
  auto take = [&](size_t bytes) { char* p = ws + off; off += (bytes + 255) & ~(size_t)255; return p; };
  bf16_t* x_bf  = (bf16_t*)take((size_t)kB * kT * kD * 2);     // 33.5 MB
  bf16_t* Wx    = (bf16_t*)take((size_t)4 * kH * kD * 2);      // 1.0 MB
  bf16_t* Uw    = (bf16_t*)take((size_t)5 * kH * kH * 2);      // 2.6 MB
  bf16_t* fcw   = (bf16_t*)take((size_t)kFC * kH * 2);
  bf16_t* hb0   = (bf16_t*)take((size_t)kB * kH * 2);
  bf16_t* hb1   = (bf16_t*)take((size_t)kB * kH * 2);
  bf16_t* cb0   = (bf16_t*)take((size_t)kB * kH * 2);
  bf16_t* cb1   = (bf16_t*)take((size_t)kB * kH * 2);
  float*  c32   = (float*)take((size_t)kB * kH * 4);
  float*  fco   = (float*)take((size_t)kB * kFC * 4);

  auto cvt = [&](const float* s, bf16_t* d, int n) {
    int blocks = (n + 255) / 256; if (blocks > 4096) blocks = 4096;
    tlstm_cvt_bf16<<<blocks, 256, 0, stream>>>(s, d, n);
  };

  // 1) bf16 conversions
  cvt(x, x_bf, kB * kT * kD);
  cvt(Wi_w, Wx + (size_t)0 * kH * kD, kH * kD);
  cvt(Wf_w, Wx + (size_t)1 * kH * kD, kH * kD);
  cvt(Wo_w, Wx + (size_t)2 * kH * kD, kH * kD);
  cvt(Wc_w, Wx + (size_t)3 * kH * kD, kH * kD);
  cvt(Ui_w, Uw + (size_t)0 * kH * kH, kH * kH);
  cvt(Uf_w, Uw + (size_t)1 * kH * kH, kH * kH);
  cvt(Uo_w, Uw + (size_t)2 * kH * kH, kH * kH);
  cvt(Uc_w, Uw + (size_t)3 * kH * kH, kH * kH);
  cvt(Wd_w, Uw + (size_t)4 * kH * kH, kH * kH);
  cvt(fc_w, fcw, kFC * kH);

  // 2) zero initial state (h0, c0) — buffer 0 is read at t=0
  tlstm_zero_state<<<(kB * kH + 255) / 256, 256, 0, stream>>>(hb0, cb0, c32, kB * kH);

  // 3) recurrent scan: 256 stream-ordered step kernels (graph-capture safe)
  StepArgs sa;
  sa.xb = x_bf; sa.timep = timep; sa.Wx = Wx; sa.Uw = Uw;
  sa.Wi_b = Wi_b; sa.bi = bi; sa.Wf_b = Wf_b; sa.bfg = bfg;
  sa.Wo_b = Wo_b; sa.bo = bo; sa.Wc_b = Wc_b; sa.bc = bc;
  sa.Ui_b = Ui_b; sa.Uf_b = Uf_b; sa.Uo_b = Uo_b; sa.Uc_b = Uc_b;
  sa.b_decomp = b_dec;
  sa.hbuf0 = hb0; sa.hbuf1 = hb1; sa.cbuf0 = cb0; sa.cbuf1 = cb1; sa.c32 = c32;
  for (int t = 0; t < kT; ++t)
    tlstm_step<<<32, 256, 0, stream>>>(sa, t, t & 1);
  // after 256 steps the final h lives in buffer 0 (write = 1 - (255&1) = 0)

  // 4) classifier head
  tlstm_fc<<<32, 256, 0, stream>>>(hb0, fcw, fc_b, fco);
  tlstm_cls<<<(kB * kO + 255) / 256, 256, 0, stream>>>(fco, cls_w, cls_b, (float*)d_out);
}
